// MultiHeadAttention_25512105738418
// MI455X (gfx1250) — matrix operand order repaired and checked by reading, not compiled, not run on hardware
//
#include <hip/hip_runtime.h>
#include <cstdint>

typedef __attribute__((ext_vector_type(16))) _Float16 v16h;
typedef __attribute__((ext_vector_type(8)))  _Float16 v8h;
typedef __attribute__((ext_vector_type(8)))  float    v8f;
typedef __attribute__((ext_vector_type(4)))  float    v4f;

#define D_MODEL 1024
#define SEQ     4096
#define HEADS   16
#define DEPTH   64

// ---------------------------------------------------------------- helpers ---

__device__ inline v16h pack16(v8h lo, v8h hi) {
  v16h r;
#pragma unroll
  for (int i = 0; i < 8; ++i) { r[i] = lo[i]; r[8 + i] = hi[i]; }
  return r;
}

__device__ inline v16h cvt16(v4f a0, v4f a1, v4f a2, v4f a3) {
  v16h r;
#pragma unroll
  for (int i = 0; i < 4; ++i) {
    r[i]      = (_Float16)a0[i];
    r[4 + i]  = (_Float16)a1[i];
    r[8 + i]  = (_Float16)a2[i];
    r[12 + i] = (_Float16)a3[i];
  }
  return r;
}

// async copy of 16 bytes global -> LDS (CDNA5, tracked by ASYNCcnt)
__device__ inline void async_copy16(void* lds_ptr, const void* gptr) {
  uint32_t lds_off = (uint32_t)(uintptr_t)lds_ptr;  // low 32 bits = LDS offset
  asm volatile("global_load_async_to_lds_b128 %0, %1, off"
               :: "v"(lds_off), "v"(gptr) : "memory");
}
__device__ inline void async_wait0() {
  asm volatile("s_wait_asynccnt 0x0" ::: "memory");
}

// ------------------------------------------------- kernel 1: QKV projection --
// y = X @ W + b   (X:[4096,1024]f32, W:[1024,1024]f32) -> f16 head-major out.
// mode 0/1: out[h][s][d] (Qh/Kh, Q scaled by 1/8);  mode 2: out[h][d][s] (Vt)
__global__ __launch_bounds__(256)
void proj_kernel(const float* __restrict__ X, const float* __restrict__ W,
                 const float* __restrict__ bias, _Float16* __restrict__ out,
                 int mode, float scale)
{
  __shared__ __align__(16) float    wraw[32 * 64];   // raw W tile f32
  __shared__ __align__(16) _Float16 wt[64 * 32];     // transposed f16: wt[n][k]

  const int tid  = threadIdx.x;
  const int lane = tid & 31;
  const int wave = tid >> 5;            // 0..7
  const int l16  = lane & 15;
  const int half = lane >> 4;
  const int m0   = blockIdx.y * 128;
  const int n0   = blockIdx.x * 64;
  const int mw   = m0 + wave * 16;      // this wave's 16 output rows

  v8f acc[4] = {};

  for (int k0 = 0; k0 < D_MODEL; k0 += 32) {
    // --- async-stage W[k0..k0+31][n0..n0+63] (8KB) into LDS ---------------
    int c = tid;
#pragma unroll
    for (int rep = 0; rep < 2; ++rep) {
      int row  = c >> 4;            // 16 x 16B chunks per 64-float row
      int col4 = (c & 15) * 4;
      async_copy16(&wraw[c * 4], W + (size_t)(k0 + row) * D_MODEL + n0 + col4);
      c += 256;
    }
    async_wait0();
    __syncthreads();
    // --- convert + transpose into wt[n][k] --------------------------------
#pragma unroll
    for (int i = 0; i < 8; ++i) {
      int e  = tid + 256 * i;       // 0..2047
      int kk = e >> 6;
      int n  = e & 63;
      wt[n * 32 + kk] = (_Float16)wraw[e];
    }
    __syncthreads();
    // --- A fragment: X row (f32 -> f16) -----------------------------------
    const float* xr = X + (size_t)(mw + l16) * D_MODEL + k0 + 8 * half;
    v16h a = cvt16(*(const v4f*)xr, *(const v4f*)(xr + 4),
                   *(const v4f*)(xr + 16), *(const v4f*)(xr + 20));
    // --- 4 WMMAs over the 64-wide N tile ----------------------------------
#pragma unroll
    for (int s = 0; s < 4; ++s) {
      const _Float16* bp = &wt[(s * 16 + l16) * 32 + half * 8];
      v16h b = pack16(*(const v8h*)bp, *(const v8h*)(bp + 16));
      acc[s] = __builtin_amdgcn_wmma_f32_16x16x32_f16(
          false, a, false, b, (short)0, acc[s], false, false);
    }
    __syncthreads();
  }

  // --- epilogue: bias, scale, head-major f16 store ------------------------
#pragma unroll
  for (int s = 0; s < 4; ++s) {
    int   ncol = n0 + s * 16 + l16;
    int   h = ncol >> 6, d = ncol & 63;
    float bv = bias[ncol];
#pragma unroll
    for (int r = 0; r < 8; ++r) {
      int    mrow = mw + r + 8 * half;
      float  val  = (acc[s][r] + bv) * scale;
      size_t off  = (mode == 2)
          ? ((size_t)h * DEPTH + d) * SEQ + mrow      // Vt[h][d][s]
          : ((size_t)h * SEQ + mrow) * DEPTH + d;     // Qh/Kh[h][s][d]
      out[off] = (_Float16)val;
    }
  }
}

// ----------------------------------------------- kernel 2: fused attention --
// one block = one head x one 16-row q tile. Scores live in LDS (16x4100 f32).
__global__ __launch_bounds__(256)
void attn_kernel(const _Float16* __restrict__ Qh, const _Float16* __restrict__ Kh,
                 const _Float16* __restrict__ Vt, float* __restrict__ attn,
                 _Float16* __restrict__ concat)
{
  extern __shared__ __align__(16) float smem[];
  const int SLD = 4100;                       // row stride (bank-split halves)
  float* S       = smem;                      // 16*4100
  float* vacc    = S + 16 * SLD;              // 16*64
  float* red     = vacc + 16 * 64;            // 16*16
  float* rowstat = red + 16 * 16;             // [0..15]=max, [16..31]=1/sum

  const int tid  = threadIdx.x;
  const int lane = tid & 31;
  const int wave = __builtin_amdgcn_readfirstlane(tid >> 5);
  const int l16  = lane & 15;
  const int half = lane >> 4;
  const int q0   = blockIdx.x * 16;
  const int h    = blockIdx.y;

  const _Float16* Qb = Qh + (size_t)h * SEQ * DEPTH;
  const _Float16* Kb = Kh + (size_t)h * SEQ * DEPTH;
  const _Float16* Vb = Vt + (size_t)h * DEPTH * SEQ;

  // Q fragments for both K-steps (d0 = 0, 32), reused across all 256 N tiles
  v16h aq[2];
#pragma unroll
  for (int ks = 0; ks < 2; ++ks) {
    const _Float16* p = Qb + (size_t)(q0 + l16) * DEPTH + ks * 32 + 8 * half;
    aq[ks] = pack16(*(const v8h*)p, *(const v8h*)(p + 16));
  }

  // ---- phase 1: scores = (Q/8) @ K^T into LDS ----------------------------
  for (int nt = wave; nt < 256; nt += 8) {
    int n0 = nt * 16;
    v8f acc = {};
#pragma unroll
    for (int ks = 0; ks < 2; ++ks) {
      const _Float16* bp = Kb + (size_t)(n0 + l16) * DEPTH + ks * 32 + half * 8;
      v16h b = pack16(*(const v8h*)bp, *(const v8h*)(bp + 16));
      acc = __builtin_amdgcn_wmma_f32_16x16x32_f16(
          false, aq[ks], false, b, (short)0, acc, false, false);
    }
#pragma unroll
    for (int r = 0; r < 8; ++r)
      S[(r + 8 * half) * SLD + n0 + l16] = acc[r];
  }
  __syncthreads();

  // ---- phase 2: softmax (16 threads per row) -----------------------------
  const int row = tid >> 4;
  const int j   = tid & 15;
  float m = -3.0e38f;
  for (int c = j * 4; c < SEQ; c += 64) {
    v4f v = *(const v4f*)&S[row * SLD + c];
    m = fmaxf(m, fmaxf(fmaxf(v[0], v[1]), fmaxf(v[2], v[3])));
  }
  red[row * 16 + j] = m;
  __syncthreads();
  if (j == 0) {
    float mm = red[row * 16];
#pragma unroll
    for (int i = 1; i < 16; ++i) mm = fmaxf(mm, red[row * 16 + i]);
    rowstat[row] = mm;
  }
  __syncthreads();
  float rmax = rowstat[row];
  float sum  = 0.f;
  for (int c = j * 4; c < SEQ; c += 64) {
    float* p = &S[row * SLD + c];
    v4f v = *(const v4f*)p;
#pragma unroll
    for (int i = 0; i < 4; ++i) { v[i] = __expf(v[i] - rmax); sum += v[i]; }
    *(v4f*)p = v;
  }
  red[row * 16 + j] = sum;
  __syncthreads();
  if (j == 0) {
    float ss = 0.f;
#pragma unroll
    for (int i = 0; i < 16; ++i) ss += red[row * 16 + i];
    rowstat[16 + row] = 1.0f / ss;
  }
  __syncthreads();
  float rinv = rowstat[16 + row];

  // ---- phase 3: normalize in LDS + single attn write; zero vacc ----------
  float* attn_row = attn + ((size_t)h * SEQ + (q0 + row)) * SEQ;
  for (int c = j * 4; c < SEQ; c += 64) {
    float* p = &S[row * SLD + c];
    v4f v = *(const v4f*)p;
#pragma unroll
    for (int i = 0; i < 4; ++i) v[i] *= rinv;
    *(v4f*)p = v;
    *(v4f*)(attn_row + c) = v;                 // global_store_b128
  }
#pragma unroll
  for (int i = 0; i < 4; ++i) vacc[tid + 256 * i] = 0.f;
  __syncthreads();

  // ---- phase 4: P @ V (each wave owns a 512-wide K slice) ----------------
  v8f oacc[4] = {};
  const int kbeg = wave * 512;
  for (int kc = kbeg; kc < kbeg + 512; kc += 32) {
    const float* ap = &S[l16 * SLD + kc + 8 * half];
    v16h a = cvt16(*(const v4f*)ap, *(const v4f*)(ap + 4),
                   *(const v4f*)(ap + 16), *(const v4f*)(ap + 20));
#pragma unroll
    for (int s = 0; s < 4; ++s) {
      const _Float16* bp = Vb + (size_t)(s * 16 + l16) * SEQ + kc + half * 8;
      v16h b = pack16(*(const v8h*)bp, *(const v8h*)(bp + 16));
      oacc[s] = __builtin_amdgcn_wmma_f32_16x16x32_f16(
          false, a, false, b, (short)0, oacc[s], false, false);
    }
  }
#pragma unroll
  for (int s = 0; s < 4; ++s)
#pragma unroll
    for (int r = 0; r < 8; ++r)
      atomicAdd(&vacc[(r + 8 * half) * 64 + s * 16 + l16], oacc[s][r]);
  __syncthreads();

  // ---- phase 5: write concat f16 [s][1024] -------------------------------
#pragma unroll
  for (int i = 0; i < 4; ++i) {
    int e = tid + 256 * i;               // 0..1023 -> 16x64 tile
    int r = e >> 6, d = e & 63;
    concat[(size_t)(q0 + r) * D_MODEL + h * 64 + d] = (_Float16)vacc[e];
  }
}

// --------------------------------------------- kernel 3: output projection --
__global__ __launch_bounds__(256)
void outproj_kernel(const _Float16* __restrict__ Xh, const float* __restrict__ W,
                    const float* __restrict__ bias, float* __restrict__ out)
{
  __shared__ __align__(16) float    wraw[32 * 64];
  __shared__ __align__(16) _Float16 wt[64 * 32];

  const int tid  = threadIdx.x;
  const int lane = tid & 31;
  const int wave = tid >> 5;
  const int l16  = lane & 15;
  const int half = lane >> 4;
  const int m0   = blockIdx.y * 128;
  const int n0   = blockIdx.x * 64;
  const int mw   = m0 + wave * 16;

  v8f acc[4] = {};

  for (int k0 = 0; k0 < D_MODEL; k0 += 32) {
    int c = tid;
#pragma unroll
    for (int rep = 0; rep < 2; ++rep) {
      int row  = c >> 4;
      int col4 = (c & 15) * 4;
      async_copy16(&wraw[c * 4], W + (size_t)(k0 + row) * D_MODEL + n0 + col4);
      c += 256;
    }
    async_wait0();
    __syncthreads();
#pragma unroll
    for (int i = 0; i < 8; ++i) {
      int e  = tid + 256 * i;
      int kk = e >> 6;
      int n  = e & 63;
      wt[n * 32 + kk] = (_Float16)wraw[e];
    }
    __syncthreads();
    const _Float16* xr = Xh + (size_t)(mw + l16) * D_MODEL + k0 + 8 * half;
    v16h a = pack16(*(const v8h*)xr, *(const v8h*)(xr + 16));
#pragma unroll
    for (int s = 0; s < 4; ++s) {
      const _Float16* bp = &wt[(s * 16 + l16) * 32 + half * 8];
      v16h b = pack16(*(const v8h*)bp, *(const v8h*)(bp + 16));
      acc[s] = __builtin_amdgcn_wmma_f32_16x16x32_f16(
          false, a, false, b, (short)0, acc[s], false, false);
    }
    __syncthreads();
  }

#pragma unroll
  for (int s = 0; s < 4; ++s) {
    int   ncol = n0 + s * 16 + l16;
    float bv = bias[ncol];
#pragma unroll
    for (int r = 0; r < 8; ++r) {
      int mrow = mw + r + 8 * half;
      out[(size_t)mrow * D_MODEL + ncol] = acc[s][r] + bv;
    }
  }
}

// ---------------------------------------------------------------- launcher --
extern "C" void kernel_launch(void* const* d_in, const int* in_sizes, int n_in,
                              void* d_out, int out_size, void* d_ws, size_t ws_size,
                              hipStream_t stream)
{
  const float* q  = (const float*)d_in[0];
  const float* k  = (const float*)d_in[1];
  const float* v  = (const float*)d_in[2];
  const float* Wq = (const float*)d_in[3];
  const float* bq = (const float*)d_in[4];
  const float* Wk = (const float*)d_in[5];
  const float* bk = (const float*)d_in[6];
  const float* Wv = (const float*)d_in[7];
  const float* bv = (const float*)d_in[8];
  const float* Wo = (const float*)d_in[9];
  const float* bo = (const float*)d_in[10];

  float* out  = (float*)d_out;                       // [4096,1024]
  float* attn = out + (size_t)SEQ * D_MODEL;         // [16,4096,4096]

  char* ws = (char*)d_ws;                            // 32 MB used
  _Float16* Qh = (_Float16*)(ws);                    // [16][4096][64]  8MB
  _Float16* Kh = (_Float16*)(ws + ((size_t)8  << 20));
  _Float16* Vt = (_Float16*)(ws + ((size_t)16 << 20)); // [16][64][4096]
  _Float16* Cc = (_Float16*)(ws + ((size_t)24 << 20)); // concat [4096][1024]

  dim3 gp(D_MODEL / 64, SEQ / 128);
  proj_kernel<<<gp, 256, 0, stream>>>(q, Wq, bq, Qh, 0, 0.125f); // 1/sqrt(64)
  proj_kernel<<<gp, 256, 0, stream>>>(k, Wk, bk, Kh, 1, 1.0f);
  proj_kernel<<<gp, 256, 0, stream>>>(v, Wv, bv, Vt, 2, 1.0f);

  size_t smem = (size_t)(16 * 4100 + 16 * 64 + 16 * 16 + 32) * sizeof(float);
  (void)hipFuncSetAttribute((const void*)attn_kernel,
                            hipFuncAttributeMaxDynamicSharedMemorySize,
                            (int)smem);
  attn_kernel<<<dim3(SEQ / 16, HEADS), 256, smem, stream>>>(Qh, Kh, Vt, attn, Cc);

  outproj_kernel<<<gp, 256, 0, stream>>>(Cc, Wo, bo, out);
}
